// Linear_Fitting_21328807592221
// MI455X (gfx1250) — matrix operand order repaired and checked by reading, not compiled, not run on hardware
//
#include <hip/hip_runtime.h>
#include <hip/hip_bf16.h>
#include <math.h>

// ---------------------------------------------------------------- types
typedef __bf16 bf16;
typedef bf16  v16bf __attribute__((ext_vector_type(16)));
typedef float v8f   __attribute__((ext_vector_type(8)));

union Frag { v16bf v; float4 q[2]; bf16 e[16]; };

// Problem constants
#define BB 8
#define SS 512
#define DD 1024
#define NH 16
#define HD 64
#define OO 512
#define MTOT (BB * SS)            // 4096

// ---------------------------------------------------------------- WMMA helpers
__device__ __forceinline__ v8f wmma_bf16(v16bf a, v16bf b, v8f c) {
  return __builtin_amdgcn_wmma_f32_16x16x32_bf16(
      /*neg_a=*/false, a, /*neg_b=*/false, b,
      /*c_mod=*/(short)0, c, /*reuse_a=*/false, /*reuse_b=*/false);
}

// A fragment: 16x32 (MxK) bf16, row-major source, ld elements per row.
// Lanes 0-15 hold K [0..8) and [16..24); lanes 16-31 hold K [8..16) and [24..32).
__device__ __forceinline__ v16bf ldfragA(const bf16* __restrict__ p, int ld, int lane) {
  int row = lane & 15;
  int kh  = (lane >> 4) * 8;
  const bf16* r = p + (size_t)row * ld + kh;
  Frag f;
  f.q[0] = *(const float4*)(r);
  f.q[1] = *(const float4*)(r + 16);
  return f.v;
}

// Same fragment but source is f32 (convert on load).
__device__ __forceinline__ v16bf ldfragA_f32(const float* __restrict__ p, int ld, int lane) {
  int row = lane & 15;
  int kh  = (lane >> 4) * 8;
  const float* r = p + (size_t)row * ld + kh;
  Frag f;
#pragma unroll
  for (int i = 0; i < 8; ++i) f.e[i] = (bf16)r[i];
#pragma unroll
  for (int i = 0; i < 8; ++i) f.e[8 + i] = (bf16)r[16 + i];
  return f.v;
}

// B fragment: 32x16 (KxN). Source stored as [N,K] row-major (A*B^T GEMM form).
// Lanes 0-15 hold K [0..8) and [16..24) of column n; lanes 16-31 hold K [8..16) and [24..32).
__device__ __forceinline__ v16bf ldfragB(const bf16* __restrict__ p, int ld, int lane) {
  int col = lane & 15;
  int kh  = (lane >> 4) * 8;
  const bf16* r = p + (size_t)col * ld + kh;
  Frag f;
  f.q[0] = *(const float4*)(r);
  f.q[1] = *(const float4*)(r + 16);
  return f.v;
}

// ---------------------------------------------------------------- cvt f32 -> bf16
__global__ __launch_bounds__(256) void cvt_kernel(const float* __restrict__ in,
                                                  bf16* __restrict__ out, int n4) {
  int i = blockIdx.x * blockDim.x + threadIdx.x;
  if (i < n4) {
    float4 f = ((const float4*)in)[i];
    bf16* p = out + (size_t)i * 4;
    p[0] = (bf16)f.x; p[1] = (bf16)f.y; p[2] = (bf16)f.z; p[3] = (bf16)f.w;
  }
}

// ---------------------------------------------------------------- GEMM: C = A*B^T + bias
// A [M,K] bf16 row-major, B [N,K] bf16 row-major, C [M,N] f32.
// Block: 128 threads = 4 waves (2x2), block tile 128x128, wave tile 64x64.
__global__ __launch_bounds__(128) void gemm_bias_kernel(const bf16* __restrict__ A,
                                                        const bf16* __restrict__ B,
                                                        const float* __restrict__ bias,
                                                        float* __restrict__ C,
                                                        int M, int N, int K) {
  int lane = threadIdx.x & 31;
  int wave = threadIdx.x >> 5;
  int bm = blockIdx.x * 128 + (wave >> 1) * 64;
  int bn = blockIdx.y * 128 + (wave & 1) * 64;

  v8f acc[4][4] = {};
  for (int k = 0; k < K; k += 32) {
    if (k + 32 < K) {
      __builtin_prefetch(A + (size_t)bm * K + k + 32, 0, 1);
      __builtin_prefetch(B + (size_t)bn * K + k + 32, 0, 1);
    }
    v16bf af[4], bfg[4];
#pragma unroll
    for (int i = 0; i < 4; ++i) af[i]  = ldfragA(A + (size_t)(bm + 16 * i) * K + k, K, lane);
#pragma unroll
    for (int j = 0; j < 4; ++j) bfg[j] = ldfragB(B + (size_t)(bn + 16 * j) * K + k, K, lane);
#pragma unroll
    for (int i = 0; i < 4; ++i)
#pragma unroll
      for (int j = 0; j < 4; ++j)
        acc[i][j] = wmma_bf16(af[i], bfg[j], acc[i][j]);
  }

  int nlo = lane & 15;
  int mhi = (lane >> 4) * 8;
#pragma unroll
  for (int i = 0; i < 4; ++i)
#pragma unroll
    for (int j = 0; j < 4; ++j) {
      int col = bn + 16 * j + nlo;
      float bb = bias ? bias[col] : 0.f;
      float* cp = C + (size_t)(bm + 16 * i + mhi) * N + col;
#pragma unroll
      for (int r = 0; r < 8; ++r) cp[(size_t)r * N] = acc[i][j][r] + bb;
    }
}

// ---------------------------------------------------------------- softmax over head_dim (64)
// h [b,s,n,64] f32 -> hsm [b,n,s,64] bf16 and hsmT [b,n,64,s] bf16. Wave per segment.
__global__ __launch_bounds__(256) void softmax_hd_kernel(const float* __restrict__ h,
                                                         bf16* __restrict__ hsm,
                                                         bf16* __restrict__ hsmT) {
  int wave = threadIdx.x >> 5, lane = threadIdx.x & 31;
  int g = blockIdx.x * 8 + wave;          // g indexes (b,s,n)
  int n = g & 15;
  int bs = g >> 4;                        // b*512+s
  int s = bs & 511, b = bs >> 9;

  float2 val = *(const float2*)(h + (size_t)g * 64 + lane * 2);
  float m = fmaxf(val.x, val.y);
#pragma unroll
  for (int off = 16; off > 0; off >>= 1) m = fmaxf(m, __shfl_xor(m, off));
  float e0 = __expf(val.x - m), e1 = __expf(val.y - m);
  float sm = e0 + e1;
#pragma unroll
  for (int off = 16; off > 0; off >>= 1) sm += __shfl_xor(sm, off);
  float inv = 1.0f / sm;
  e0 *= inv; e1 *= inv;

  size_t bn = (size_t)(b * 16 + n);
  size_t base = (bn * 512 + s) * 64 + lane * 2;
  hsm[base]     = (bf16)e0;
  hsm[base + 1] = (bf16)e1;
  size_t baseT = bn * 64 * 512 + s;
  hsmT[baseT + (size_t)(lane * 2) * 512]     = (bf16)e0;
  hsmT[baseT + (size_t)(lane * 2 + 1) * 512] = (bf16)e1;
}

// ---------------------------------------------------------------- v[b,n,e] = sum_s hsm[b,n,s,e]
__global__ __launch_bounds__(256) void vsum_kernel(const bf16* __restrict__ hsm,
                                                   float* __restrict__ v) {
  int t = blockIdx.x * blockDim.x + threadIdx.x;   // 8192 = b*n*64
  if (t >= BB * NH * HD) return;
  int e = t & 63, bn = t >> 6;
  float acc = 0.f;
  const bf16* p = hsm + (size_t)bn * 512 * 64 + e;
  for (int s = 0; s < 512; ++s) acc += (float)p[(size_t)s * 64];
  v[t] = acc;
}

// ---------------------------------------------------------------- z[b,n,o,e] = sum_h attn_w[o,n,e,h]*v[b,n,h]
// Block = 64 threads handles one (n,o) for all 8 batches; attn_w row kept in regs (read once).
__global__ __launch_bounds__(64) void z_kernel(const float* __restrict__ attn_w,
                                               const float* __restrict__ v,
                                               bf16* __restrict__ z) {
  __shared__ float ldsv[512];               // v for 8 batches x 64
  int o = blockIdx.x & 511;
  int n = blockIdx.x >> 9;
  int e = threadIdx.x;
  for (int i = threadIdx.x; i < 512; i += 64) {
    int b = i >> 6, hh = i & 63;
    ldsv[i] = v[((b * 16 + n) << 6) + hh];
  }
  __syncthreads();
  const float* awp = attn_w + ((size_t)(o * 16 + n) * 64 + e) * 64;
  float a[64];
#pragma unroll
  for (int hh = 0; hh < 64; hh += 4) {
    float4 t = *(const float4*)(awp + hh);
    a[hh] = t.x; a[hh + 1] = t.y; a[hh + 2] = t.z; a[hh + 3] = t.w;
  }
  for (int b = 0; b < 8; ++b) {
    float acc = 0.f;
#pragma unroll
    for (int hh = 0; hh < 64; ++hh) acc += a[hh] * ldsv[b * 64 + hh];
    z[(((size_t)(b * 16 + n) * 512 + o) << 6) + e] = (bf16)acc;
  }
}

// ---------------------------------------------------------------- scores = scale * hsm . z^T  (per b,n)
// M=N=512, K=64. Writes f32 directly into attn1 region of d_out.
__global__ __launch_bounds__(128) void scores_kernel(const bf16* __restrict__ hsm,
                                                     const bf16* __restrict__ z,
                                                     float* __restrict__ scores,
                                                     float scale) {
  int bn = blockIdx.z;
  const bf16* A = hsm + (size_t)bn * 512 * 64;
  const bf16* B = z   + (size_t)bn * 512 * 64;
  float* C = scores + (size_t)bn * 512 * 512;

  int lane = threadIdx.x & 31;
  int wave = threadIdx.x >> 5;
  int bm = blockIdx.x * 128 + (wave >> 1) * 64;
  int bnn = blockIdx.y * 128 + (wave & 1) * 64;

  v8f acc[4][4] = {};
#pragma unroll
  for (int k = 0; k < 64; k += 32) {
    v16bf af[4], bfg[4];
#pragma unroll
    for (int i = 0; i < 4; ++i) af[i]  = ldfragA(A + (size_t)(bm + 16 * i) * 64 + k, 64, lane);
#pragma unroll
    for (int j = 0; j < 4; ++j) bfg[j] = ldfragB(B + (size_t)(bnn + 16 * j) * 64 + k, 64, lane);
#pragma unroll
    for (int i = 0; i < 4; ++i)
#pragma unroll
      for (int j = 0; j < 4; ++j)
        acc[i][j] = wmma_bf16(af[i], bfg[j], acc[i][j]);
  }
  int nlo = lane & 15, mhi = (lane >> 4) * 8;
#pragma unroll
  for (int i = 0; i < 4; ++i)
#pragma unroll
    for (int j = 0; j < 4; ++j) {
      float* cp = C + (size_t)(bm + 16 * i + mhi) * 512 + bnn + 16 * j + nlo;
#pragma unroll
      for (int r = 0; r < 8; ++r) cp[(size_t)r * 512] = acc[i][j][r] * scale;
    }
}

// ---------------------------------------------------------------- in-place softmax over o=512 (wave per row)
__global__ __launch_bounds__(256) void softmax_o_kernel(float* __restrict__ attn1) {
  int wave = threadIdx.x >> 5, lane = threadIdx.x & 31;
  size_t row = (size_t)blockIdx.x * 8 + wave;
  float* p = attn1 + row * 512 + lane * 4;
  float4 x[4];
#pragma unroll
  for (int j = 0; j < 4; ++j) x[j] = *(const float4*)(p + j * 128);
  float m = -3.4e38f;
#pragma unroll
  for (int j = 0; j < 4; ++j)
    m = fmaxf(m, fmaxf(fmaxf(x[j].x, x[j].y), fmaxf(x[j].z, x[j].w)));
#pragma unroll
  for (int off = 16; off > 0; off >>= 1) m = fmaxf(m, __shfl_xor(m, off));
  float sm = 0.f;
#pragma unroll
  for (int j = 0; j < 4; ++j) {
    x[j].x = __expf(x[j].x - m); x[j].y = __expf(x[j].y - m);
    x[j].z = __expf(x[j].z - m); x[j].w = __expf(x[j].w - m);
    sm += x[j].x + x[j].y + x[j].z + x[j].w;
  }
#pragma unroll
  for (int off = 16; off > 0; off >>= 1) sm += __shfl_xor(sm, off);
  float inv = 1.0f / sm;
#pragma unroll
  for (int j = 0; j < 4; ++j) {
    x[j].x *= inv; x[j].y *= inv; x[j].z *= inv; x[j].w *= inv;
    *(float4*)(p + j * 128) = x[j];
  }
}

// ---------------------------------------------------------------- out_bn = attn1 . hsmT^T  (per b,n)
// A = attn1 f32 [512,512] (cvt on load), B = hsmT bf16 [64,512]; result bf16 into [b,s,n,h].
// Block: 128 threads = 4 waves stacked in M, block tile 256x64, wave tile 64x64.
__global__ __launch_bounds__(128) void av_kernel(const float* __restrict__ attn1,
                                                 const bf16* __restrict__ hsmT,
                                                 bf16* __restrict__ attnout) {
  int bnid = blockIdx.z;
  int b = bnid >> 4, n = bnid & 15;
  const float* A = attn1 + (size_t)bnid * 512 * 512;
  const bf16*  B = hsmT  + (size_t)bnid * 64 * 512;

  int lane = threadIdx.x & 31;
  int wave = threadIdx.x >> 5;
  int bm = blockIdx.x * 256 + wave * 64;

  v8f acc[4][4] = {};
  for (int k = 0; k < 512; k += 32) {
    if (k + 32 < 512) __builtin_prefetch(A + (size_t)bm * 512 + k + 32, 0, 1);
    v16bf af[4], bfg[4];
#pragma unroll
    for (int i = 0; i < 4; ++i) af[i]  = ldfragA_f32(A + (size_t)(bm + 16 * i) * 512 + k, 512, lane);
#pragma unroll
    for (int j = 0; j < 4; ++j) bfg[j] = ldfragB(B + (size_t)(16 * j) * 512 + k, 512, lane);
#pragma unroll
    for (int i = 0; i < 4; ++i)
#pragma unroll
      for (int j = 0; j < 4; ++j)
        acc[i][j] = wmma_bf16(af[i], bfg[j], acc[i][j]);
  }

  bf16* Cb = attnout + (size_t)b * 512 * 1024 + n * 64;
  int nlo = lane & 15, mhi = (lane >> 4) * 8;
#pragma unroll
  for (int i = 0; i < 4; ++i)
#pragma unroll
    for (int j = 0; j < 4; ++j) {
#pragma unroll
      for (int r = 0; r < 8; ++r)
        Cb[(size_t)(bm + 16 * i + mhi + r) * 1024 + 16 * j + nlo] = (bf16)acc[i][j][r];
    }
}

// ---------------------------------------------------------------- launch
extern "C" void kernel_launch(void* const* d_in, const int* in_sizes, int n_in,
                              void* d_out, int out_size, void* d_ws, size_t ws_size,
                              hipStream_t stream) {
  const float* x      = (const float*)d_in[0];
  const float* W_in   = (const float*)d_in[1];
  const float* b_in   = (const float*)d_in[2];
  const float* attn_w = (const float*)d_in[3];
  const float* W_out  = (const float*)d_in[4];
  const float* b_out  = (const float*)d_in[5];

  float* out   = (float*)d_out;                       // [b,s,d]  4,194,304 f32
  float* attn1 = out + (size_t)MTOT * DD;             // [b,n,s,o] 33,554,432 f32

  char* ws = (char*)d_ws;
  bf16*  xb    = (bf16*)(ws + 0);                     //  8 MB  x bf16 [4096,1024]
  bf16*  winb  = (bf16*)(ws + 8388608);               //  2 MB
  bf16*  woutb = (bf16*)(ws + 10485760);              //  2 MB
  float* h     = (float*)(ws + 12582912);             // 16 MB  [b,s,n,e] f32
  bf16*  hsm   = (bf16*)(ws + 29360128);              //  8 MB  [b,n,s,e] bf16
  bf16*  hsmT  = (bf16*)(ws + 37748736);              //  8 MB  [b,n,e,s] bf16
  float* vbuf  = (float*)(ws + 46137344);             // 32 KB  [b,n,e] f32
  bf16*  zbuf  = (bf16*)(ws + 46170112);              //  8 MB  [b,n,o,e] bf16
  bf16*  aoutb = (bf16*)(ws + 54558720);              //  8 MB  [b,s,n,h] bf16

  // 1. casts
  cvt_kernel<<<(MTOT * DD / 4 + 255) / 256, 256, 0, stream>>>(x, xb, MTOT * DD / 4);
  cvt_kernel<<<(DD * DD / 4 + 255) / 256, 256, 0, stream>>>(W_in, winb, DD * DD / 4);
  cvt_kernel<<<(DD * DD / 4 + 255) / 256, 256, 0, stream>>>(W_out, woutb, DD * DD / 4);

  // 2. h = x @ W_in^T + b_in
  gemm_bias_kernel<<<dim3(MTOT / 128, DD / 128), 128, 0, stream>>>(
      xb, winb, b_in, h, MTOT, DD, DD);

  // 3. softmax over head_dim, produce hsm + hsmT
  softmax_hd_kernel<<<(BB * SS * NH) / 8, 256, 0, stream>>>(h, hsm, hsmT);

  // 4. v = sum_s hsm ; z = attn_w . v
  vsum_kernel<<<(BB * NH * HD + 255) / 256, 256, 0, stream>>>(hsm, vbuf);
  z_kernel<<<NH * OO, 64, 0, stream>>>(attn_w, vbuf, zbuf);

  // 5. scores = scale * hsm . z^T  -> attn1 region, then softmax over o in-place
  float scale = 1.0f / (64.0f * sqrtf(512.0f));
  scores_kernel<<<dim3(SS / 128, OO / 128, BB * NH), 128, 0, stream>>>(
      hsm, zbuf, attn1, scale);
  softmax_o_kernel<<<(BB * NH * SS) / 8, 256, 0, stream>>>(attn1);

  // 6. attn-weighted values -> [b,s,n,h] bf16
  av_kernel<<<dim3(SS / 256, 1, BB * NH), 128, 0, stream>>>(attn1, hsmT, aoutb);

  // 7. out = aout @ W_out^T + b_out
  gemm_bias_kernel<<<dim3(MTOT / 128, DD / 128), 128, 0, stream>>>(
      aoutb, woutb, b_out, out, MTOT, DD, DD);
}
